// LSTM_9586367005060
// MI455X (gfx1250) — compile-verified, matrix operand order checked by reading, not run on hardware
//
#include <hip/hip_runtime.h>

#define TT 512
#define BB 128
#define II 512
#define HH 1024

typedef __attribute__((ext_vector_type(16))) __bf16 v16bf;
typedef __attribute__((ext_vector_type(8)))  float  v8f;

// ---------------- helpers ----------------

__device__ __forceinline__ unsigned short f2bf_bits(float f) {
  unsigned u = __builtin_bit_cast(unsigned, f);
  u += 0x7FFFu + ((u >> 16) & 1u);          // round-to-nearest-even
  return (unsigned short)(u >> 16);
}

__device__ __forceinline__ __bf16 f2bf(float f) {
  unsigned short h = f2bf_bits(f);
  return __builtin_bit_cast(__bf16, h);
}

__device__ __forceinline__ float fast_sigmoid(float x) {
  return 1.0f / (1.0f + __expf(-x));
}

// tanh(x) = 2*sigmoid(2x) - 1   (no NaN at large |x|: exp->inf -> 0 -> -1)
__device__ __forceinline__ float fast_tanh(float x) {
  return 2.0f / (1.0f + __expf(-2.0f * x)) - 1.0f;
}

__device__ __forceinline__ v8f wmma_bf16(v16bf a, v16bf b, v8f c) {
  return __builtin_amdgcn_wmma_f32_16x16x32_bf16(
      /*neg_a=*/false, a, /*neg_b=*/false, b,
      /*c_mod=*/(short)0, c, /*reuse_a=*/false, /*reuse_b=*/false);
}

// ---------------- conversion / transpose kernels ----------------

// X (fp32) -> bf16, 4 elements per thread, packed 8-byte stores
__global__ __launch_bounds__(256) void cvt_f32_bf16(const float* __restrict__ in,
                                                    __bf16* __restrict__ out,
                                                    long n) {
  long i = ((long)blockIdx.x * blockDim.x + threadIdx.x) * 4;
  if (i + 3 < n) {
    float4 v = *(const float4*)(in + i);
    unsigned lo = (unsigned)f2bf_bits(v.x) | ((unsigned)f2bf_bits(v.y) << 16);
    unsigned hi = (unsigned)f2bf_bits(v.z) | ((unsigned)f2bf_bits(v.w) << 16);
    uint2 pk; pk.x = lo; pk.y = hi;
    *(uint2*)(out + i) = pk;
  }
}

// W[K][N] fp32 -> WT[N][K] bf16 (coalesced writes; one-time cost)
__global__ __launch_bounds__(256) void transpose_w_bf16(const float* __restrict__ W,
                                                        __bf16* __restrict__ WT,
                                                        int K, int N) {
  long i = (long)blockIdx.x * blockDim.x + threadIdx.x;
  if (i >= (long)K * N) return;
  int n = (int)(i / K);
  int k = (int)(i % K);
  WT[i] = f2bf(W[(long)k * N + n]);
}

// ---------------- fused LSTM step kernel ----------------
//
// grid = 32 blocks x 128 threads (4 waves). 128 waves total.
// wave w owns rows [m0, m0+32) of B and h-columns [n0, n0+32) of H,
// computing z for all 4 gates (a,i,f,o) -> 16 wmma output tiles.
//
// A (16x32 bf16, k order used here): lane L<16 holds row M=L, K=0..15; L>=16 holds row M=L-16, K=16..31.
// B (32x16 bf16, same k order as A): lane L<16 holds col N=L, K=0..15; L>=16 holds col N=L-16, K=16..31.
//   -> weights pre-transposed to [gate][N][K] so each lane loads 16 contiguous K values.
// C/D (16x16 f32): VGPR r, lane L: M = (L>=16)*8 + r, N = L%16.

__global__ __launch_bounds__(128) void lstm_step(
    const __bf16* __restrict__ Xb,     // [T,B,I] bf16
    const __bf16* __restrict__ WxT,    // [4,H,I] bf16
    const __bf16* __restrict__ WhT,    // [4,H,H] bf16
    const __bf16* __restrict__ hb_in,  // [B,H] bf16 (valid when t>0)
    __bf16* __restrict__ hb_out,       // [B,H] bf16
    const float* __restrict__ ba, const float* __restrict__ bi,
    const float* __restrict__ bfg, const float* __restrict__ bo,
    const float* s_in,                 // [B,H] fp32 (read when t>0)
    float* s_out,                      // [B,H] fp32 (may alias s_in)
    float* __restrict__ out,           // [T,B,H] fp32 (d_out base)
    int t) {
  const int lane  = threadIdx.x & 31;
  const int wave  = threadIdx.x >> 5;
  const int w     = blockIdx.x * 4 + wave;   // 0..127
  const int m0    = (w & 3) * 32;            // batch-row block
  const int n0    = (w >> 2) * 32;           // h-column block
  const int col   = lane & 15;
  const int khalf = (lane >> 4) * 16;

  v8f acc[4][2][2];
  const v8f zacc = {0.f, 0.f, 0.f, 0.f, 0.f, 0.f, 0.f, 0.f};
#pragma unroll
  for (int g = 0; g < 4; ++g)
#pragma unroll
    for (int ti = 0; ti < 2; ++ti)
#pragma unroll
      for (int tj = 0; tj < 2; ++tj) acc[g][ti][tj] = zacc;

  // ---- X_t @ Wx  (K = II) ----
  {
    const __bf16* a0p = Xb + ((size_t)t * BB + (size_t)(m0 + col)) * II + khalf;
    const __bf16* a1p = a0p + (size_t)16 * II;
    const __bf16* bp[4][2];
#pragma unroll
    for (int g = 0; g < 4; ++g)
#pragma unroll
      for (int tj = 0; tj < 2; ++tj)
        bp[g][tj] = WxT + ((size_t)g * HH + (size_t)(n0 + tj * 16 + col)) * II + khalf;

    for (int k = 0; k < II; k += 32) {
      v16bf a0 = *(const v16bf*)(a0p + k);
      v16bf a1 = *(const v16bf*)(a1p + k);
#pragma unroll
      for (int g = 0; g < 4; ++g)
#pragma unroll
        for (int tj = 0; tj < 2; ++tj) {
          v16bf b = *(const v16bf*)(bp[g][tj] + k);
          acc[g][0][tj] = wmma_bf16(a0, b, acc[g][0][tj]);
          acc[g][1][tj] = wmma_bf16(a1, b, acc[g][1][tj]);
        }
    }
  }

  // ---- h_{t-1} @ Wh  (K = HH) ----
  if (t > 0) {
    const __bf16* a0p = hb_in + (size_t)(m0 + col) * HH + khalf;
    const __bf16* a1p = a0p + (size_t)16 * HH;
    const __bf16* bp[4][2];
#pragma unroll
    for (int g = 0; g < 4; ++g)
#pragma unroll
      for (int tj = 0; tj < 2; ++tj)
        bp[g][tj] = WhT + ((size_t)g * HH + (size_t)(n0 + tj * 16 + col)) * HH + khalf;

    for (int k = 0; k < HH; k += 32) {
      v16bf a0 = *(const v16bf*)(a0p + k);
      v16bf a1 = *(const v16bf*)(a1p + k);
#pragma unroll
      for (int g = 0; g < 4; ++g)
#pragma unroll
        for (int tj = 0; tj < 2; ++tj) {
          v16bf b = *(const v16bf*)(bp[g][tj] + k);
          acc[g][0][tj] = wmma_bf16(a0, b, acc[g][0][tj]);
          acc[g][1][tj] = wmma_bf16(a1, b, acc[g][1][tj]);
        }
    }
  }

  // ---- fused gate activations + cell/hidden update (all in registers) ----
  float bav[2], biv[2], bfv[2], bov[2];
#pragma unroll
  for (int tj = 0; tj < 2; ++tj) {
    const int n = n0 + tj * 16 + col;
    bav[tj] = ba[n]; biv[tj] = bi[n]; bfv[tj] = bfg[n]; bov[tj] = bo[n];
  }

  const size_t outbase = (size_t)t * BB * HH;
#pragma unroll
  for (int ti = 0; ti < 2; ++ti) {
#pragma unroll
    for (int tj = 0; tj < 2; ++tj) {
      const int n  = n0 + tj * 16 + col;
      const int mb = m0 + ti * 16 + (lane >> 4) * 8;
#pragma unroll
      for (int r = 0; r < 8; ++r) {
        const size_t off = (size_t)(mb + r) * HH + (size_t)n;
        float za = acc[0][ti][tj][r] + bav[tj];
        float zi = acc[1][ti][tj][r] + biv[tj];
        float zf = acc[2][ti][tj][r] + bfv[tj];
        float zo = acc[3][ti][tj][r] + bov[tj];
        float ag = fast_tanh(za);
        float ig = fast_sigmoid(zi);
        float fg = fast_sigmoid(zf);
        float og = fast_sigmoid(zo);
        float sp = (t == 0) ? 0.0f : s_in[off];
        float sn = ag * ig + sp * fg;
        float hn = fast_tanh(sn) * og;
        s_out[off]          = sn;
        out[outbase + off]  = hn;
        hb_out[off]         = f2bf(hn);
      }
    }
  }
}

// ---------------- host-side launcher ----------------

extern "C" void kernel_launch(void* const* d_in, const int* in_sizes, int n_in,
                              void* d_out, int out_size, void* d_ws, size_t ws_size,
                              hipStream_t stream) {
  (void)in_sizes; (void)n_in; (void)out_size; (void)ws_size;

  const float* X     = (const float*)d_in[0];
  const float* Wx[4] = {(const float*)d_in[1], (const float*)d_in[2],
                        (const float*)d_in[3], (const float*)d_in[4]};
  const float* Wh[4] = {(const float*)d_in[5], (const float*)d_in[6],
                        (const float*)d_in[7], (const float*)d_in[8]};
  const float* ba = (const float*)d_in[9];
  const float* bi = (const float*)d_in[10];
  const float* bf = (const float*)d_in[11];
  const float* bo = (const float*)d_in[12];

  float* out     = (float*)d_out;
  float* s_final = out + (size_t)TT * BB * HH;   // tail of d_out: s_last [B,H]

  // workspace carve-up (~78 MB)
  char* p = (char*)d_ws;
  __bf16* Xb  = (__bf16*)p; p += (size_t)TT * BB * II * sizeof(__bf16);  // 64 MB
  __bf16* WxT = (__bf16*)p; p += (size_t)4 * HH * II * sizeof(__bf16);   //  4 MB
  __bf16* WhT = (__bf16*)p; p += (size_t)4 * HH * HH * sizeof(__bf16);   //  8 MB
  float*  sbuf = (float*)p; p += (size_t)BB * HH * sizeof(float);        // 0.5 MB
  __bf16* hb0 = (__bf16*)p; p += (size_t)BB * HH * sizeof(__bf16);
  __bf16* hb1 = (__bf16*)p;
  __bf16* hb[2] = {hb0, hb1};

  // 1) X fp32 -> bf16  (n = 33,554,432; 4 per thread; exact division)
  {
    const long n = (long)TT * BB * II;
    const int  threads = 256;
    const int  blocks  = (int)(n / (4 * threads));
    cvt_f32_bf16<<<blocks, threads, 0, stream>>>(X, Xb, n);
  }

  // 2) weight transpose+convert: Wx [I,H] -> WxT[g][H][I]; Wh [H,H] -> WhT[g][H][H]
  for (int g = 0; g < 4; ++g) {
    const long nx = (long)HH * II;
    transpose_w_bf16<<<(int)((nx + 255) / 256), 256, 0, stream>>>(
        Wx[g], WxT + (size_t)g * HH * II, II, HH);
    const long nh = (long)HH * HH;
    transpose_w_bf16<<<(int)((nh + 255) / 256), 256, 0, stream>>>(
        Wh[g], WhT + (size_t)g * HH * HH, HH, HH);
  }

  // 3) recurrence: one fused WMMA step kernel per timestep.
  //    h_t is written both to d_out (fp32 outputs) and a bf16 ping-pong buffer.
  for (int t = 0; t < TT; ++t) {
    const __bf16* hin  = hb[(t + 1) & 1];             // h_{t-1} (unused at t=0)
    __bf16*       hout = hb[t & 1];
    float*        sout = (t == TT - 1) ? s_final : sbuf;
    lstm_step<<<32, 128, 0, stream>>>(Xb, WxT, WhT, hin, hout,
                                      ba, bi, bf, bo,
                                      sbuf, sout, out, t);
  }
}
